// Decoder_62483184222858
// MI455X (gfx1250) — matrix operand order repaired and checked by reading, not compiled, not run on hardware
//
#include <hip/hip_runtime.h>

#define EMBED   512
#define HIDDEN  1024
#define VOCAB   1004
#define VOCABP  1024
#define BATCH   512
#define TLEN    65
#define NSTEP   64
#define G4      4096   // 4*HIDDEN

typedef __attribute__((ext_vector_type(16))) __bf16 bf16x16;
typedef __attribute__((ext_vector_type(8)))  float  floatx8;

union FragU { bf16x16 v; uint4 q[2]; };

// A-matrix 16x32 bf16 fragment (ISA 7.12.2): lane L -> row (L%16);
// lanes 0-15: K = [col..col+7] and [col+16..col+23]; lanes 16-31: +8.
__device__ __forceinline__ bf16x16 load_fragA(const __bf16* __restrict__ p,
                                              int ld, int row, int col) {
  const int lane = threadIdx.x & 31;
  const __bf16* base = p + (size_t)(row + (lane & 15)) * ld + col + ((lane >> 4) << 3);
  FragU f;
  f.q[0] = *reinterpret_cast<const uint4*>(base);
  f.q[1] = *reinterpret_cast<const uint4*>(base + 16);
  return f.v;
}

// B-matrix 32x16 bf16 fragment: lane L -> column (L%16) of B (= row of W),
// K = (L/16)*8 + [0..7] and [16..23]. W stored [N][K] row-major == B^T rows.
__device__ __forceinline__ bf16x16 load_fragB(const __bf16* __restrict__ p,
                                              int ld, int row, int col) {
  const int lane = threadIdx.x & 31;
  const __bf16* base = p + (size_t)(row + (lane & 15)) * ld + col + ((lane >> 4) << 3);
  FragU f;
  f.q[0] = *reinterpret_cast<const uint4*>(base);
  f.q[1] = *reinterpret_cast<const uint4*>(base + 16);
  return f.v;
}

__device__ __forceinline__ floatx8 wmma_bf16(bf16x16 a, bf16x16 b, floatx8 c) {
  return __builtin_amdgcn_wmma_f32_16x16x32_bf16(false, a, false, b, (short)0, c,
                                                 false, false);
}

// ---------------- prep kernels ----------------
__global__ void cast_f32_bf16(const float* __restrict__ src, __bf16* __restrict__ dst, int n) {
  int i = blockIdx.x * blockDim.x + threadIdx.x;
  if (i < n) dst[i] = (__bf16)src[i];
}

__global__ void zero_f32(float* __restrict__ p, int n) {
  int i = blockIdx.x * blockDim.x + threadIdx.x;
  if (i < n) p[i] = 0.f;
}

__global__ void bias_sum(const float* __restrict__ bih, const float* __restrict__ bhh,
                         float* __restrict__ bias, int n) {
  int i = blockIdx.x * blockDim.x + threadIdx.x;
  if (i < n) bias[i] = bih[i] + bhh[i];
}

// lin_W [1004,1024] -> bf16 [1024,1024] zero-padded rows
__global__ void cast_linW(const float* __restrict__ src, __bf16* __restrict__ dst) {
  int i = blockIdx.x * blockDim.x + threadIdx.x;  // over 1024*1024
  int r = i >> 10, c = i & 1023;
  float v = (r < VOCAB) ? src[r * HIDDEN + c] : 0.f;
  dst[i] = (__bf16)v;
}

__global__ void pad_linb(const float* __restrict__ src, float* __restrict__ dst) {
  int i = blockIdx.x * blockDim.x + threadIdx.x;
  if (i < VOCABP) dst[i] = (i < VOCAB) ? src[i] : 0.f;
}

// X[t][b][e] = bf16(embed_W[captions[b][t]][e]), t in [0,64)
__global__ void embed_gather(const int* __restrict__ captions,
                             const float* __restrict__ embW,
                             __bf16* __restrict__ X) {
  int i = blockIdx.x * blockDim.x + threadIdx.x;  // 64*512*512 = 16M
  int e = i & (EMBED - 1);
  int b = (i >> 9) & (BATCH - 1);
  int t = i >> 18;
  int tok = captions[b * TLEN + t];
  X[i] = (__bf16)embW[(size_t)tok * EMBED + e];
}

// ---------------- recurrent step ----------------
// grid (16,16), block 256 (8 waves). Workgroup tile: 32 rows x 64 units,
// all 4 gates. wave = (row_half, gate). K = 512 (x) + 1024 (h).
__launch_bounds__(256)
__global__ void lstm_step(const __bf16* __restrict__ Xt,      // [512][512]
                          const __bf16* __restrict__ Hprev,   // [512][1024]
                          const __bf16* __restrict__ Wih,     // [4096][512]
                          const __bf16* __restrict__ Whh,     // [4096][1024]
                          const float*  __restrict__ bias,    // [4096]
                          const float*  __restrict__ feature, // [512][1024]
                          float*        __restrict__ C,       // [512][1024]
                          __bf16*       __restrict__ Hout) {  // [512][1024]
  __shared__ float lds[4][32][68];  // [gate][row][unit], padded
  const int lane = threadIdx.x & 31;
  const int wave = threadIdx.x >> 5;
  const int gate = wave & 3;
  const int rh   = wave >> 2;                // 0/1
  const int row0 = blockIdx.x * 32 + rh * 16;
  const int ub   = blockIdx.y * 64;

  floatx8 acc[4] = {};

  for (int kk = 0; kk < EMBED; kk += 32) {
    bf16x16 a = load_fragA(Xt, EMBED, row0, kk);
#pragma unroll
    for (int j = 0; j < 4; ++j) {
      bf16x16 b = load_fragB(Wih, EMBED, gate * HIDDEN + ub + j * 16, kk);
      acc[j] = wmma_bf16(a, b, acc[j]);
    }
  }
  for (int kk = 0; kk < HIDDEN; kk += 32) {
    bf16x16 a = load_fragA(Hprev, HIDDEN, row0, kk);
#pragma unroll
    for (int j = 0; j < 4; ++j) {
      bf16x16 b = load_fragB(Whh, HIDDEN, gate * HIDDEN + ub + j * 16, kk);
      acc[j] = wmma_bf16(a, b, acc[j]);
    }
  }

#pragma unroll
  for (int j = 0; j < 4; ++j) {
    const int ucol = ub + j * 16 + (lane & 15);
    const float bv = bias[gate * HIDDEN + ucol];
#pragma unroll
    for (int v = 0; v < 8; ++v) {
      const int r = rh * 16 + v + ((lane >> 4) << 3);
      lds[gate][r][j * 16 + (lane & 15)] = acc[j][v] + bv;
    }
  }
  __syncthreads();

  const int rowBase = blockIdx.x * 32;
  for (int idx = threadIdx.x; idx < 32 * 64; idx += 256) {
    const int r = idx >> 6, u = idx & 63;
    const size_t off = (size_t)(rowBase + r) * HIDDEN + (ub + u);
    const float iv = 1.f / (1.f + __expf(-lds[0][r][u]));
    const float fv = 1.f / (1.f + __expf(-lds[1][r][u]));
    const float gv = tanhf(lds[2][r][u]);
    const float ov = 1.f / (1.f + __expf(-lds[3][r][u]));
    const float cn = fv * C[off] + iv * gv;
    const float hn = ov * tanhf(cn) + feature[off];
    C[off] = cn;
    Hout[off] = (__bf16)hn;
  }
}

// ---------------- final projection ----------------
// O rows are t-major (m = t*512 + b); output is [b][t][v].
// grid (1024, 4), block 256. Workgroup tile 32 rows x 256 cols.
__launch_bounds__(256)
__global__ void logits_gemm(const __bf16* __restrict__ O,    // [32768][1024]
                            const __bf16* __restrict__ Wl,   // [1024][1024] padded
                            const float*  __restrict__ bl,   // [1024] padded
                            float*        __restrict__ out) {// [512][64][1004]
  const int lane = threadIdx.x & 31;
  const int wave = threadIdx.x >> 5;
  const int ch   = wave & 3;
  const int rh   = wave >> 2;
  const int row0 = blockIdx.x * 32 + rh * 16;
  const int col0 = blockIdx.y * 256 + ch * 64;

  floatx8 acc[4] = {};
  for (int kk = 0; kk < HIDDEN; kk += 32) {
    bf16x16 a = load_fragA(O, HIDDEN, row0, kk);
#pragma unroll
    for (int j = 0; j < 4; ++j) {
      bf16x16 b = load_fragB(Wl, HIDDEN, col0 + j * 16, kk);
      acc[j] = wmma_bf16(a, b, acc[j]);
    }
  }
#pragma unroll
  for (int j = 0; j < 4; ++j) {
    const int n = col0 + j * 16 + (lane & 15);
    if (n < VOCAB) {
      const float bv = bl[n];
#pragma unroll
      for (int v = 0; v < 8; ++v) {
        const int m = row0 + v + ((lane >> 4) << 3);
        const int t = m >> 9;        // rows are t-major
        const int b = m & 511;
        out[((size_t)b * NSTEP + t) * VOCAB + n] = acc[j][v] + bv;
      }
    }
  }
}

extern "C" void kernel_launch(void* const* d_in, const int* in_sizes, int n_in,
                              void* d_out, int out_size, void* d_ws, size_t ws_size,
                              hipStream_t stream) {
  const float* feature = (const float*)d_in[0];
  const int*   caps    = (const int*)d_in[1];
  // d_in[2] lengths: unused by reference output
  const float* embW    = (const float*)d_in[3];
  const float* W_ih    = (const float*)d_in[4];
  const float* W_hh    = (const float*)d_in[5];
  const float* b_ih    = (const float*)d_in[6];
  const float* b_hh    = (const float*)d_in[7];
  const float* lin_W   = (const float*)d_in[8];
  const float* lin_b   = (const float*)d_in[9];
  float* out = (float*)d_out;

  char* ws = (char*)d_ws;
  auto alloc = [&](size_t bytes) -> char* {
    char* p = ws;
    ws += (bytes + 255) & ~(size_t)255;
    return p;
  };
  const size_t BH = (size_t)BATCH * HIDDEN;                       // 524288
  __bf16* Xemb  = (__bf16*)alloc((size_t)NSTEP * BATCH * EMBED * 2); // 32 MB
  __bf16* WihB  = (__bf16*)alloc((size_t)G4 * EMBED * 2);            // 4 MB
  __bf16* WhhB  = (__bf16*)alloc((size_t)G4 * HIDDEN * 2);           // 8 MB
  __bf16* WlB   = (__bf16*)alloc((size_t)VOCABP * HIDDEN * 2);       // 2 MB
  __bf16* Fbf   = (__bf16*)alloc(BH * 2);                            // 1 MB
  float*  bias  = (float*)alloc((size_t)G4 * 4);
  float*  blp   = (float*)alloc((size_t)VOCABP * 4);
  float*  Cst   = (float*)alloc(BH * 4);                             // 2 MB
  __bf16* Obf   = (__bf16*)alloc((size_t)NSTEP * BH * 2);            // 64 MB

  const int TB = 256;
  cast_f32_bf16<<<(G4 * EMBED) / TB, TB, 0, stream>>>(W_ih, WihB, G4 * EMBED);
  cast_f32_bf16<<<(G4 * HIDDEN) / TB, TB, 0, stream>>>(W_hh, WhhB, G4 * HIDDEN);
  cast_f32_bf16<<<(int)(BH / TB), TB, 0, stream>>>(feature, Fbf, (int)BH);
  cast_linW<<<(VOCABP * HIDDEN) / TB, TB, 0, stream>>>(lin_W, WlB);
  bias_sum<<<G4 / TB, TB, 0, stream>>>(b_ih, b_hh, bias, G4);
  pad_linb<<<VOCABP / TB, TB, 0, stream>>>(lin_b, blp);
  zero_f32<<<(int)(BH / TB), TB, 0, stream>>>(Cst, (int)BH);
  embed_gather<<<(NSTEP * BATCH * EMBED) / TB, TB, 0, stream>>>(caps, embW, Xemb);

  for (int t = 0; t < NSTEP; ++t) {
    const __bf16* Hprev = (t == 0) ? Fbf : (Obf + (size_t)(t - 1) * BH);
    lstm_step<<<dim3(BATCH / 32, HIDDEN / 64), TB, 0, stream>>>(
        Xemb + (size_t)t * BATCH * EMBED, Hprev, WihB, WhhB, bias, feature,
        Cst, Obf + (size_t)t * BH);
  }

  logits_gemm<<<dim3((NSTEP * BATCH) / 32, VOCABP / 256), TB, 0, stream>>>(
      Obf, WlB, blp, out);
}
